// graphRCNNFeatureExtractor_17222818857603
// MI455X (gfx1250) — matrix operand order repaired and checked by reading, not compiled, not run on hardware
//
#include <hip/hip_runtime.h>

// ---------------------------------------------------------------------------
// MI455X (gfx1250, wave32) implementation.
// GEMMs run on v_wmma_f32_16x16x32_f16 (f16 in, f32 accumulate).
// node2edge gathers are fused into the GEMM A-tile loader (segment+index),
// weights are pre-packed into native WMMA B-fragment layout.
// ---------------------------------------------------------------------------

typedef __attribute__((ext_vector_type(16))) _Float16 v16h;
typedef __attribute__((ext_vector_type(8)))  _Float16 v8h;
typedef __attribute__((ext_vector_type(8)))  float    v8f;

enum : int {
  kNKP = 14, kNE = 182, kB = 1024, kH = 256,
  kME = kB * kNE,      // 186368 edge rows  (divisible by 128)
  kMN = kB * kNKP      // 14336 node rows   (divisible by 128)
};

// Input index map (d_in): [0]=x, [1]=ratio, then params leaves in JAX
// pytree order (dict keys sorted): dec(14 leaves), enc(26), enc3d(26), mean_shape.
enum : int {
  I_X = 0, I_RATIO = 1,
  I_DEC_M1W0 = 2, I_DEC_M1B0 = 3, I_DEC_M1W1 = 4, I_DEC_M1B1 = 5,
  I_DEC_M2W0 = 6, I_DEC_M2B0 = 7, I_DEC_M2W1 = 8, I_DEC_M2B1 = 9,
  I_DEC_O1W = 10, I_DEC_O1B = 11, I_DEC_O2W = 12, I_DEC_O2B = 13,
  I_DEC_O3W = 14, I_DEC_O3B = 15,
  I_ENC = 16,    // fc_out.W,fc_out.b, then mlp1..4: beta,fc1.W,fc1.b,fc2.W,fc2.b,gamma
  I_ENC3D = 42,
  I_MS = 68
};

struct Seg { const _Float16* p; const int* idx; int ld; int w; };

__device__ __forceinline__ v16h mk16(v8h lo, v8h hi) {
  return __builtin_shufflevector(lo, hi, 0,1,2,3,4,5,6,7,8,9,10,11,12,13,14,15);
}

// --------------------------- big GEMM (N == 256) ---------------------------
// C(M x 256, f16) = act(A @ Wpacked + bias). A is up to 3 K-segments, each with
// optional per-row gather index (fused node2edge). Block 128x128, 8 waves
// (2Mx4N), each wave 64x32 = 8 WMMA tiles. K step 32 via LDS-staged A.
template<int ACT>  // 1 = ELU, 2 = ReLU
__global__ void __launch_bounds__(256)
gemm_big(Seg s0, Seg s1, Seg s2,
         const _Float16* __restrict__ Bp, const float* __restrict__ bias,
         _Float16* __restrict__ out, int N, int nkt)
{
  __shared__ _Float16 lA[128 * 40];           // 128 rows x 32 halves, pad to 40
  const int tid  = threadIdx.x;
  const int lane = tid & 31;
  const int wid  = tid >> 5;
  const int wN = wid & 3, wM = wid >> 2;
  const int ml = lane & 15;
  const int abase = (lane & 16) ? 8 : 0;      // ISA 16-bit A frag k-base per lane-half
  const int nT = N >> 4;
  v8f acc[4][2] = {};
  const int w01 = s0.w + s1.w;

  for (int kt = 0; kt < nkt; ++kt) {
    const int k0 = kt << 5;
    const _Float16* sp; const int* sidx; int sld, kl;
    if (k0 < s0.w)     { sp = s0.p; sidx = s0.idx; sld = s0.ld; kl = k0; }
    else if (k0 < w01) { sp = s1.p; sidx = s1.idx; sld = s1.ld; kl = k0 - s0.w; }
    else               { sp = s2.p; sidx = s2.idx; sld = s2.ld; kl = k0 - w01; }
    __syncthreads();
    {
      const int r  = tid >> 1;
      const int hs = (tid & 1) << 4;
      const int grow = blockIdx.x * 128 + r;
      const int srow = sidx ? sidx[grow] : grow;      // fused gather
      const v8h* src = (const v8h*)(sp + (size_t)srow * sld + kl + hs);
      v8h d0 = src[0], d1 = src[1];
      *(v8h*)(lA + r * 40 + hs)     = d0;
      *(v8h*)(lA + r * 40 + hs + 8) = d1;
    }
    __syncthreads();
    const int ntb = blockIdx.y * 8 + wN * 2;
    v16h bf0 = *(const v16h*)(Bp + (((size_t)kt * nT + ntb    ) << 9) + (lane << 4));
    v16h bf1 = *(const v16h*)(Bp + (((size_t)kt * nT + ntb + 1) << 9) + (lane << 4));
#pragma unroll
    for (int ms = 0; ms < 4; ++ms) {
      const _Float16* rp = lA + (wM * 64 + ms * 16 + ml) * 40;
      v8h lo = *(const v8h*)(rp + abase);
      v8h hi = *(const v8h*)(rp + abase + 16);
      v16h af = mk16(lo, hi);
      acc[ms][0] = __builtin_amdgcn_wmma_f32_16x16x32_f16(false, af, false, bf0,
                     (short)0, acc[ms][0], false, false);
      acc[ms][1] = __builtin_amdgcn_wmma_f32_16x16x32_f16(false, af, false, bf1,
                     (short)0, acc[ms][1], false, false);
    }
  }
  const int rOff = (lane & 16) ? 8 : 0;
#pragma unroll
  for (int ms = 0; ms < 4; ++ms)
#pragma unroll
  for (int ns = 0; ns < 2; ++ns) {
    const int gn = blockIdx.y * 128 + wN * 32 + ns * 16 + ml;
    const float bv = bias[gn];
#pragma unroll
    for (int r = 0; r < 8; ++r) {
      const size_t gm = (size_t)blockIdx.x * 128 + wM * 64 + ms * 16 + r + rOff;
      float v = acc[ms][ns][r] + bv;
      if (ACT == 1) v = v > 0.f ? v : (__expf(v) - 1.f);
      else          v = v > 0.f ? v : 0.f;
      out[gm * N + gn] = (_Float16)v;
    }
  }
}

// ------------------- skinny GEMM (N padded to 16, f32 out) -----------------
template<int ADDX>
__global__ void __launch_bounds__(256)
gemm_skinny(const _Float16* __restrict__ A, int lda,
            const _Float16* __restrict__ Bp, const float* __restrict__ bias,
            const float* __restrict__ addsrc, int addld,
            float* __restrict__ out, int ldout, int Nvalid, int nkt)
{
  const int lane = threadIdx.x & 31;
  const int wid  = threadIdx.x >> 5;
  const size_t m0 = (size_t)blockIdx.x * 128 + wid * 16;
  const int ml = lane & 15;
  const int abase = (lane & 16) ? 8 : 0;
  v8f acc = {};
  const _Float16* arow = A + (m0 + ml) * (size_t)lda;
  for (int kt = 0; kt < nkt; ++kt) {
    const _Float16* rp = arow + (kt << 5);
    v8h lo = *(const v8h*)(rp + abase);
    v8h hi = *(const v8h*)(rp + abase + 16);
    v16h af = mk16(lo, hi);
    v16h bf = *(const v16h*)(Bp + ((size_t)kt << 9) + (lane << 4));
    acc = __builtin_amdgcn_wmma_f32_16x16x32_f16(false, af, false, bf,
            (short)0, acc, false, false);
  }
  if (ml < Nvalid) {
    const float bv = bias[ml];
    const int rOff = (lane & 16) ? 8 : 0;
#pragma unroll
    for (int r = 0; r < 8; ++r) {
      const size_t gm = m0 + r + rOff;
      float v = acc[r] + bv;
      if (ADDX) v += addsrc[gm * addld + ml];
      out[gm * ldout + ml] = v;
    }
  }
}

// ---------------------------- helper kernels -------------------------------
__global__ void build_idx_k(int* __restrict__ srow, int* __restrict__ rrow) {
  int i = blockIdx.x * 256 + threadIdx.x;
  if (i >= kME) return;
  int b = i / kNE, e = i % kNE;
  int snd = e / 13, jj = e % 13;
  int rcv = jj + (jj >= snd ? 1 : 0);
  srow[i] = b * kNKP + snd;
  rrow[i] = b * kNKP + rcv;
}

// Pack f32 weight (Ksrc x Nsrc, row-major) into WMMA B-fragment layout
// [(kt*nT+nt)*32 + lane]*16 + ei, with k = kt*32 + (lane>=16?8:0) + (ei&7) + (ei>=8?16:0),
// n = nt*16 + lane%16. split/seg0/gap implement the out_fc1 K-remap.
__global__ void pack_w_k(const float* __restrict__ W, _Float16* __restrict__ out,
                         int Ksrc, int Nsrc, int Kp, int Np,
                         int split, int seg0, int gap, int total) {
  int i = blockIdx.x * 256 + threadIdx.x;
  if (i >= total) return;
  int ei = i & 15;
  int lane = (i >> 4) & 31;
  int tile = i >> 9;
  int nT = Np >> 4;
  int nt = tile % nT, kt = tile / nT;
  int k = kt * 32 + ((lane & 16) ? 8 : 0) + (ei & 7) + ((ei & 8) ? 16 : 0);
  int n = nt * 16 + (lane & 15);
  int sr = (k < split) ? k : (k - gap);
  bool ok = (k < split) ? (sr < seg0) : (sr < Ksrc);
  float v = (ok && n < Nsrc) ? W[(size_t)sr * Nsrc + n] : 0.f;
  out[i] = (_Float16)v;
}

__global__ void lin2_elu_k(const float* __restrict__ x, const float* __restrict__ W,
                           const float* __restrict__ b, _Float16* __restrict__ out) {
  int i = blockIdx.x * 256 + threadIdx.x;      // over kMN*256
  int c = i & 255, r = i >> 8;
  float v = x[r * 2] * W[c] + x[r * 2 + 1] * W[256 + c] + b[c];
  out[i] = (_Float16)(v > 0.f ? v : (__expf(v) - 1.f));
}

__global__ void msg1_k(const float* __restrict__ x, const int* __restrict__ srow,
                       const int* __restrict__ rrow, const float* __restrict__ W,
                       const float* __restrict__ b, _Float16* __restrict__ out) {
  int i = blockIdx.x * 256 + threadIdx.x;      // over kME*256
  int c = i & 255, r = i >> 8;
  int sA = srow[r], sB = rrow[r];
  float v = b[c] + x[sA * 2] * W[c] + x[sA * 2 + 1] * W[256 + c]
                 + x[sB * 2] * W[512 + c] + x[sB * 2 + 1] * W[768 + c];
  out[i] = (_Float16)(v > 0.f ? v : 0.f);
}

__global__ void xpad_k(const float* __restrict__ x, _Float16* __restrict__ out) {
  int i = blockIdx.x * 256 + threadIdx.x;
  if (i >= kMN * 32) return;
  int c = i & 31, r = i >> 5;
  out[i] = (_Float16)(c < 2 ? x[r * 2 + c] : 0.f);
}

__global__ void bn_stats_k(const _Float16* __restrict__ buf, int M,
                           float* __restrict__ mu, float* __restrict__ rstd) {
  __shared__ float s1[256], s2[256];
  int c = blockIdx.x;
  float a = 0.f, b = 0.f;
  for (int r = threadIdx.x; r < M; r += 256) {
    float v = (float)buf[((size_t)r << 8) + c];
    a += v; b += v * v;
  }
  s1[threadIdx.x] = a; s2[threadIdx.x] = b;
  __syncthreads();
  for (int st = 128; st > 0; st >>= 1) {
    if (threadIdx.x < st) { s1[threadIdx.x] += s1[threadIdx.x + st];
                            s2[threadIdx.x] += s2[threadIdx.x + st]; }
    __syncthreads();
  }
  if (threadIdx.x == 0) {
    float m = s1[0] / (float)M;
    float var = s2[0] / (float)M - m * m;
    mu[c] = m; rstd[c] = rsqrtf(var + 1e-5f);
  }
}

__global__ void bn_apply_k(_Float16* __restrict__ buf, const float* __restrict__ mu,
                           const float* __restrict__ rstd, const float* __restrict__ g,
                           const float* __restrict__ be, int total) {
  int i = blockIdx.x * 256 + threadIdx.x;
  if (i >= total) return;
  int c = i & 255;
  float v = (float)buf[i];
  buf[i] = (_Float16)((v - mu[c]) * rstd[c] * g[c] + be[c]);
}

__global__ void e2n_k(const _Float16* __restrict__ ebuf, _Float16* __restrict__ nbuf,
                      float scale) {
  int i = blockIdx.x * 256 + threadIdx.x;      // over kMN*256
  int c = i & 255, rn = i >> 8;
  int b = rn / kNKP, n = rn % kNKP;
  float acc = 0.f;
  for (int snd = 0; snd < kNKP; ++snd) {
    if (snd == n) continue;
    int e = snd * 13 + (n - (n > snd ? 1 : 0));
    acc += (float)ebuf[(((size_t)(b * kNE + e)) << 8) + c];
  }
  nbuf[i] = (_Float16)(acc * scale);
}

__global__ void combine_k(const _Float16* __restrict__ m0, const _Float16* __restrict__ m1,
                          const float* __restrict__ ed, _Float16* __restrict__ out) {
  int i = blockIdx.x * 256 + threadIdx.x;      // over kME*256
  int r = i >> 8;
  out[i] = (_Float16)((float)m0[i] * ed[2 * r] + (float)m1[i] * ed[2 * r + 1]);
}

__global__ void softmax2_k(const float* __restrict__ logits, float* __restrict__ ed) {
  int r = blockIdx.x * 256 + threadIdx.x;
  if (r >= kME) return;
  float l0 = logits[2 * r], l1 = logits[2 * r + 1];
  float m = fmaxf(l0, l1);
  float e0 = __expf(l0 - m), e1 = __expf(l1 - m);
  float inv = 1.f / (e0 + e1);
  ed[2 * r] = e0 * inv; ed[2 * r + 1] = e1 * inv;
}

__global__ void proj_k(const float* __restrict__ rt, const float* __restrict__ ratio,
                       const float* __restrict__ ms, float* __restrict__ out) {
  int b = blockIdx.x * 64 + threadIdx.x;
  if (b >= kB) return;
  const float c0 = rt[b * 12 + 0], cx = rt[b * 12 + 1], cy = rt[b * 12 + 2];
  const float q0 = rt[b * 12 + 3], q1 = rt[b * 12 + 4];
  const float q2 = rt[b * 12 + 5], q3 = rt[b * 12 + 6];
  const float rr = ratio[b];
  for (int p = 0; p < kNKP; ++p) {
    float X0 = 0.f, X1 = 0.f, X2 = 0.f;
    if (p != 8 && p != 13) {
      int pp = (p < 8) ? p : p - 1;
      int i6 = pp >> 1;
      X0 = ms[i6 * 3 + 0]; X1 = ms[i6 * 3 + 1];
      X2 = (pp & 1) ? -ms[i6 * 3 + 2] : ms[i6 * 3 + 2];
    }
    // t = hamilton((0,X), conj(q))
    const float b1 = -q1, b2 = -q2, b3 = -q3;
    const float t0 = -X0 * b1 - X1 * b2 - X2 * b3;
    const float t1 =  X0 * q0 + X1 * b3 - X2 * b2;
    const float t2 = -X0 * b3 + X1 * q0 + X2 * b1;
    const float t3 =  X0 * b2 - X1 * b1 + X2 * q0;
    // y = hamilton(q, t), take vector part
    const float y1 = q0 * t1 + q1 * t0 + q2 * t3 - q3 * t2;
    const float y2 = q0 * t2 - q1 * t3 + q2 * t0 + q3 * t1;
    const float y3 = q0 * t3 + q1 * t2 - q2 * t1 + q3 * t0;
    out[b * 42 + p * 3 + 0] = c0 * y1 + cx;
    out[b * 42 + p * 3 + 1] = (c0 * y2 + cy) * rr;
    out[b * 42 + p * 3 + 2] = c0 * y3;
  }
}

// ------------------------------- host side ---------------------------------
struct PkEnc { _Float16 *m1fc2, *m2fc1, *m2fc2, *m3fc1, *m3fc2, *m4fc1, *m4fc2, *fcout; };

extern "C" void kernel_launch(void* const* d_in, const int* in_sizes, int n_in,
                              void* d_out_, int out_size, void* d_ws, size_t ws_size,
                              hipStream_t s) {
  (void)in_sizes; (void)n_in; (void)out_size; (void)ws_size;
  float* d_out = (float*)d_out_;
  auto F = [&](int i) { return (const float*)d_in[i]; };

  size_t off = 0;
  auto alloc = [&](size_t bytes) -> void* {
    void* p = (char*)d_ws + off;
    off += (bytes + 255) & ~(size_t)255;
    return p;
  };

  _Float16* EB_A = (_Float16*)alloc((size_t)kME * kH * 2);
  _Float16* EB_B = (_Float16*)alloc((size_t)kME * kH * 2);
  _Float16* EB_S = (_Float16*)alloc((size_t)kME * kH * 2);
  _Float16* NB0  = (_Float16*)alloc((size_t)kMN * kH * 2);
  _Float16* NB1  = (_Float16*)alloc((size_t)kMN * kH * 2);
  _Float16* NAG  = (_Float16*)alloc((size_t)kMN * kH * 2);
  _Float16* XPD  = (_Float16*)alloc((size_t)kMN * 32 * 2);
  float*    EDG  = (float*)alloc((size_t)kME * 2 * 4);
  float*    RT   = (float*)alloc((size_t)kB * 12 * 4);
  float*    MU   = (float*)alloc(256 * 4);
  float*    RSTD = (float*)alloc(256 * 4);
  int*      SROW = (int*)alloc((size_t)kME * 4);
  int*      RROW = (int*)alloc((size_t)kME * 4);

  PkEnc pkE, pkE3;
  auto allocEnc = [&](PkEnc& pk, size_t fcoutEls) {
    pk.m1fc2 = (_Float16*)alloc(65536 * 2);
    pk.m2fc1 = (_Float16*)alloc(131072 * 2);
    pk.m2fc2 = (_Float16*)alloc(65536 * 2);
    pk.m3fc1 = (_Float16*)alloc(65536 * 2);
    pk.m3fc2 = (_Float16*)alloc(65536 * 2);
    pk.m4fc1 = (_Float16*)alloc(196608 * 2);
    pk.m4fc2 = (_Float16*)alloc(65536 * 2);
    pk.fcout = (_Float16*)alloc(fcoutEls * 2);
  };
  allocEnc(pkE,  (size_t)256 * 16);
  allocEnc(pkE3, (size_t)46592 * 16);
  _Float16* pkMsg0 = (_Float16*)alloc(65536 * 2);
  _Float16* pkMsg1 = (_Float16*)alloc(65536 * 2);
  _Float16* pkOut1 = (_Float16*)alloc((size_t)288 * 256 * 2);
  _Float16* pkOut2 = (_Float16*)alloc(65536 * 2);
  _Float16* pkOut3 = (_Float16*)alloc((size_t)256 * 16 * 2);

  // 0) edge gather indices
  build_idx_k<<<(kME + 255) / 256, 256, 0, s>>>(SROW, RROW);

  // 1) pack weights into WMMA B-fragment layout
  auto packW = [&](const float* W, _Float16* dst, int Ksrc, int Nsrc, int Kp, int Np,
                   int split, int seg0, int gap) {
    int total = (Kp / 32) * (Np / 16) * 512;
    pack_w_k<<<(total + 255) / 256, 256, 0, s>>>(W, dst, Ksrc, Nsrc, Kp, Np,
                                                 split, seg0, gap, total);
  };
  auto packN = [&](const float* W, _Float16* dst, int K, int Nsrc, int Np) {
    packW(W, dst, K, Nsrc, K, Np, 1 << 30, K, 0);
  };
  auto packEnc = [&](int eb, PkEnc& pk, int foK, int foN) {
    auto mb = [&](int m) { return eb + 2 + (m - 1) * 6; };
    packN(F(mb(1) + 3), pk.m1fc2, 256, 256, 256);
    packN(F(mb(2) + 1), pk.m2fc1, 512, 256, 256);
    packN(F(mb(2) + 3), pk.m2fc2, 256, 256, 256);
    packN(F(mb(3) + 1), pk.m3fc1, 256, 256, 256);
    packN(F(mb(3) + 3), pk.m3fc2, 256, 256, 256);
    packN(F(mb(4) + 1), pk.m4fc1, 768, 256, 256);
    packN(F(mb(4) + 3), pk.m4fc2, 256, 256, 256);
    packN(F(eb),        pk.fcout, foK, foN, 16);
  };
  packEnc(I_ENC,   pkE,  256,   2);
  packEnc(I_ENC3D, pkE3, 46592, 12);
  packN(F(I_DEC_M2W0), pkMsg0, 256, 256, 256);
  packN(F(I_DEC_M2W1), pkMsg1, 256, 256, 256);
  packW(F(I_DEC_O1W), pkOut1, 258, 256, 288, 256, 32, 2, 30);  // [x(2)|pad|agg(256)]
  packN(F(I_DEC_O2W), pkOut2, 256, 256, 256);
  packN(F(I_DEC_O3W), pkOut3, 256, 2, 16);

  auto segP = [](const _Float16* p, int w) { return Seg{p, nullptr, w, w}; };
  auto segI = [](const _Float16* p, const int* idx, int w) { return Seg{p, idx, w, w}; };
  Seg z{nullptr, nullptr, 0, 0};

  auto gemmE = [&](Seg a, Seg b, Seg c, const _Float16* Bp, const float* bias,
                   _Float16* out, int M, int K, int act) {
    dim3 g(M / 128, 2);
    if (act == 1) gemm_big<1><<<g, 256, 0, s>>>(a, b, c, Bp, bias, out, 256, K / 32);
    else          gemm_big<2><<<g, 256, 0, s>>>(a, b, c, Bp, bias, out, 256, K / 32);
  };
  auto bn = [&](_Float16* buf, int M, const float* gamma, const float* beta) {
    bn_stats_k<<<256, 256, 0, s>>>(buf, M, MU, RSTD);
    int total = M * 256;
    bn_apply_k<<<(total + 255) / 256, 256, 0, s>>>(buf, MU, RSTD, gamma, beta, total);
  };

  auto runEnc = [&](int eb, PkEnc& pk, float* skOut, int skLdout, int skNvalid,
                    int skM, int skLda, int skK) {
    auto mb = [&](int m) { return eb + 2 + (m - 1) * 6; };
    // mlp1: elu(2->256) VALU, elu(256->256) WMMA, BN
    lin2_elu_k<<<kMN, 256, 0, s>>>(F(I_X), F(mb(1) + 1), F(mb(1) + 2), NB0);
    gemmE(segP(NB0, 256), z, z, pk.m1fc2, F(mb(1) + 4), NB1, kMN, 256, 1);
    bn(NB1, kMN, F(mb(1) + 5), F(mb(1) + 0));
    // mlp2: fused node2edge gather (send|recv), K=512
    gemmE(segI(NB1, SROW, 256), segI(NB1, RROW, 256), z,
          pk.m2fc1, F(mb(2) + 2), EB_A, kME, 512, 1);
    gemmE(segP(EB_A, 256), z, z, pk.m2fc2, F(mb(2) + 4), EB_B, kME, 256, 1);
    bn(EB_B, kME, F(mb(2) + 5), F(mb(2) + 0));       // EB_B == skip
    // edge2node (mean over 13 incoming edges, /14)
    e2n_k<<<kMN, 256, 0, s>>>(EB_B, NB0, 1.f / 14.f);
    // mlp3
    gemmE(segP(NB0, 256), z, z, pk.m3fc1, F(mb(3) + 2), NB1, kMN, 256, 1);
    gemmE(segP(NB1, 256), z, z, pk.m3fc2, F(mb(3) + 4), NB0, kMN, 256, 1);
    bn(NB0, kMN, F(mb(3) + 5), F(mb(3) + 0));
    // mlp4: fused [send|recv|skip], K=768
    gemmE(segI(NB0, SROW, 256), segI(NB0, RROW, 256), segP(EB_B, 256),
          pk.m4fc1, F(mb(4) + 2), EB_A, kME, 768, 1);
    gemmE(segP(EB_A, 256), z, z, pk.m4fc2, F(mb(4) + 4), EB_B, kME, 256, 1);
    bn(EB_B, kME, F(mb(4) + 5), F(mb(4) + 0));
    // fc_out (skinny WMMA, N padded to 16)
    gemm_skinny<0><<<skM / 128, 256, 0, s>>>(EB_B, skLda, pk.fcout, F(eb + 1),
                                             nullptr, 0, skOut, skLdout, skNvalid,
                                             skK / 32);
  };

  // ---- enc -> logits + softmax edges ----
  runEnc(I_ENC, pkE, d_out, 2, 2, kME, 256, 256);
  softmax2_k<<<(kME + 255) / 256, 256, 0, s>>>(d_out, EDG);

  // ---- decoder -> kgnn2d ----
  msg1_k<<<kME, 256, 0, s>>>(F(I_X), SROW, RROW, F(I_DEC_M1W0), F(I_DEC_M1B0), EB_A);
  gemmE(segP(EB_A, 256), z, z, pkMsg0, F(I_DEC_M2B0), EB_B, kME, 256, 2);
  msg1_k<<<kME, 256, 0, s>>>(F(I_X), SROW, RROW, F(I_DEC_M1W1), F(I_DEC_M1B1), EB_A);
  gemmE(segP(EB_A, 256), z, z, pkMsg1, F(I_DEC_M2B1), EB_S, kME, 256, 2);
  combine_k<<<kME, 256, 0, s>>>(EB_B, EB_S, EDG, EB_A);
  e2n_k<<<kMN, 256, 0, s>>>(EB_A, NAG, 1.f);
  xpad_k<<<(kMN * 32 + 255) / 256, 256, 0, s>>>(F(I_X), XPD);
  gemmE(segP(XPD, 32), segP(NAG, 256), z, pkOut1, F(I_DEC_O1B), NB0, kMN, 288, 2);
  gemmE(segP(NB0, 256), z, z, pkOut2, F(I_DEC_O2B), NB1, kMN, 256, 2);
  gemm_skinny<1><<<kMN / 128, 256, 0, s>>>(NB1, 256, pkOut3, F(I_DEC_O3B),
                                           F(I_X), 2, d_out + (size_t)kME * 2,
                                           2, 2, 8);

  // ---- enc3d -> rt (flatten GEMM 1024 x 46592 x 12) ----
  runEnc(I_ENC3D, pkE3, RT, 12, 12, kB, 46592, 46592);

  // ---- ortho projection -> proj ----
  proj_k<<<kB / 64, 64, 0, s>>>(RT, F(I_RATIO), F(I_MS),
                                d_out + (size_t)kME * 2 + (size_t)kMN * 2);
}
